// MGUCell_44659069944296
// MI455X (gfx1250) — matrix operand order repaired and checked by reading, not compiled, not run on hardware
//
#include <hip/hip_runtime.h>
#include <hip/hip_bf16.h>

// ---------------- problem constants ----------------
#define B_    64
#define S_    512
#define IN_   256
#define H_    512
#define OUT_  256
#define KTOT  (H_ + IN_)      // 768 : combined [h, x] K dimension
#define KCH_H (H_ / 32)       // 16  : k-chunks covering the h part
#define KCH_T (KTOT / 32)     // 24  : total k-chunks per recurrent GEMM
#define KCH_O (H_ / 32)       // 16  : k-chunks for the output projection
#define RB_   16              // batch rows per recurrence workgroup (1 m-tile)

typedef __attribute__((ext_vector_type(16))) __bf16 v16bf;
typedef __attribute__((ext_vector_type(8)))  __bf16 v8bf;
typedef __attribute__((ext_vector_type(8)))  float  v8f;

__device__ __forceinline__ v8f zero8() {
    v8f z = {0.f, 0.f, 0.f, 0.f, 0.f, 0.f, 0.f, 0.f};
    return z;
}

// A fragment (16x32 bf16, row-major source, row stride ld elements).
// lane L: row = (L & 15); elements 0..7 -> K = kbase + (L>>4)*8 + i,
//         elements 8..15 -> K = kbase + 16 + (L>>4)*8 + i.
__device__ __forceinline__ v16bf load_a_frag(const __bf16* base, int ld,
                                             int row, int kbase, int lane) {
    const int koff = (lane >> 4) * 8;
    const __bf16* p0 = base + (long)row * ld + kbase + koff;
    v8bf lo = *(const v8bf*)(p0);
    v8bf hi = *(const v8bf*)(p0 + 16);
    v16bf a;
#pragma unroll
    for (int i = 0; i < 8; ++i) { a[i] = lo[i]; a[i + 8] = hi[i]; }
    return a;
}

// B fragment from pre-packed weights: one aligned 32-byte load per lane.
__device__ __forceinline__ v16bf load_b_frag(const __bf16* chunk, int lane) {
    return *(const v16bf*)(chunk + lane * 16);
}

// ---------------- prep kernels ----------------
__global__ void cvt_bf16_kernel(const float* __restrict__ src,
                                __bf16* __restrict__ dst, long n) {
    for (long i = (long)blockIdx.x * blockDim.x + threadIdx.x; i < n;
         i += (long)gridDim.x * blockDim.x)
        dst[i] = (__bf16)src[i];
}

// Pack W[K][N] (f32, row-major) into WMMA B-fragment order:
// idx = ((ntile*kchunks + kc)*32 + lane)*16 + i
// value = W[kc*32 + (i>>3)*16 + (lane>>4)*8 + (i&7)][ntile*16 + (lane&15)]
__global__ void pack_w_kernel(const float* __restrict__ W,
                              __bf16* __restrict__ P, int K, int N) {
    const int kchunks = K >> 5;
    const long total = (long)K * N;
    for (long idx = (long)blockIdx.x * blockDim.x + threadIdx.x; idx < total;
         idx += (long)gridDim.x * blockDim.x) {
        const int  i    = (int)(idx & 15);
        const int  l    = (int)((idx >> 4) & 31);
        const long rest = idx >> 9;
        const int  kc   = (int)(rest % kchunks);
        const int  tn   = (int)(rest / kchunks);
        const int  k    = kc * 32 + (i >> 3) * 16 + (l >> 4) * 8 + (i & 7);
        const int  n    = tn * 16 + (l & 15);
        P[idx] = (__bf16)W[(long)k * N + n];
    }
}

// ---------------- recurrence: 4 blocks x 16 batch rows ----------------
// The MGU step is independent per batch row, so each workgroup runs the full
// 512-step chain for its 16 rows (one m-tile). Wave w owns the 16x16 tile at
// columns [w*16, w*16+16): a single v8f accumulator + 8 gate floats per lane.
__global__ __launch_bounds__(1024) void mgu_rec_kernel(
    const __bf16* __restrict__ xbf,   // [B][S][IN] bf16
    const __bf16* __restrict__ wfp,   // packed Wf  [32][24][32][16]
    const __bf16* __restrict__ wsp,   // packed Ws  [32][24][32][16]
    const float*  __restrict__ bfv,   // [H]
    const float*  __restrict__ bsv,   // [H]
    __bf16* __restrict__ hsbf,        // [B*S][H] row = b*S + t (for projection)
    float*  __restrict__ final_out)   // [B][H] final hidden state (f32)
{
    __shared__ __bf16 h_s[RB_ * H_];  // 16 KB : this block's hidden slice
    __shared__ __bf16 g_s[RB_ * H_];  // 16 KB : f * h slice

    const int tid   = threadIdx.x;
    const int lane  = tid & 31;
    const int wave  = tid >> 5;          // 0..31 -> n-strip
    const int nbase = wave * 16;
    const int col   = lane & 15;
    const int rsel  = lane >> 4;         // 0/1 -> +0 / +8 row offset in C tile
    const int rbase = blockIdx.x * RB_;  // first batch row of this block
    const int arow  = lane & 15;         // local A row for fragment loads

    const float fb = bfv[nbase + col];
    const float sb = bsv[nbase + col];

    for (int i = tid; i < RB_ * H_; i += 1024) h_s[i] = (__bf16)0.0f;
    __syncthreads();

    float fval[8];                       // sigmoid gate across both GEMMs

    for (int t = 0; t < S_; ++t) {
        // Launder a ZERO scalar offset each iteration: the weight base
        // pointers keep their global address space (-> global_load_b128 with
        // scalar base, LOADcnt only, no FLAT/DScnt coupling), but the loads
        // can't be proven loop-invariant, so they are neither hoisted out of
        // the t-loop nor spilled to per-lane scratch.
        long lz = 0;
        asm volatile("" : "+s"(lz));
        const __bf16* wf_t = wfp + lz + (long)wave * KCH_T * 512;
        const __bf16* ws_t = wsp + lz + (long)wave * KCH_T * 512;

        const __bf16* xrow = xbf + ((long)(rbase + arow) * S_ + t) * IN_;

        // ---- GEMM1: pre_f = [h, x_t] @ Wf (this block's 16 rows) ----
        v8f acc = zero8();
        for (int kc = 0; kc < KCH_T; ++kc) {
            v16bf bfrag = load_b_frag(wf_t + (long)kc * 512, lane);
            v16bf afrag = (kc < KCH_H)
                ? load_a_frag(h_s, H_, arow, kc * 32, lane)
                : load_a_frag(xrow, IN_, 0, (kc - KCH_H) * 32, lane);
            acc = __builtin_amdgcn_wmma_f32_16x16x32_bf16(
                false, afrag, false, bfrag, (short)0, acc, false, false);
        }

        // ---- gate: f = sigmoid(pre_f + bf); g = f*h ----
#pragma unroll
        for (int r = 0; r < 8; ++r) {
            const float f = 1.0f / (1.0f + __expf(-(acc[r] + fb)));
            fval[r] = f;
            const int row = r + rsel * 8;
            const int j   = nbase + col;
            const float hv = (float)h_s[row * H_ + j];
            g_s[row * H_ + j] = (__bf16)(f * hv);
        }
        __syncthreads();   // g visible to all waves

        // ---- GEMM2: pre_s = [f*h, x_t] @ Ws ----
        acc = zero8();
        for (int kc = 0; kc < KCH_T; ++kc) {
            v16bf bfrag = load_b_frag(ws_t + (long)kc * 512, lane);
            v16bf afrag = (kc < KCH_H)
                ? load_a_frag(g_s, H_, arow, kc * 32, lane)
                : load_a_frag(xrow, IN_, 0, (kc - KCH_H) * 32, lane);
            acc = __builtin_amdgcn_wmma_f32_16x16x32_bf16(
                false, afrag, false, bfrag, (short)0, acc, false, false);
        }

        // ---- h_new = (1-f)*h + f*tanh(pre_s + bs) ----
#pragma unroll
        for (int r = 0; r < 8; ++r) {
            const float st = tanhf(acc[r] + sb);
            const float f  = fval[r];
            const int row = r + rsel * 8;
            const int j   = nbase + col;
            const float hv = (float)h_s[row * H_ + j];
            const float hn = (1.0f - f) * hv + f * st;
            const int grow = rbase + row;
            hsbf[((long)grow * S_ + t) * H_ + j] = (__bf16)hn;
            h_s[row * H_ + j] = (__bf16)hn;
            if (t == S_ - 1) final_out[(long)grow * H_ + j] = hn;
        }
        __syncthreads();   // h_new visible before next step's GEMM1
    }
}

// ---------------- parallel output projection: relu(hs @ Wo + bo) -----------
__global__ __launch_bounds__(256) void proj_kernel(
    const __bf16* __restrict__ hsbf,  // [B*S][H]
    const __bf16* __restrict__ wop,   // packed Wo [16][16][32][16]
    const float*  __restrict__ bo,
    float* __restrict__ out)          // [B*S][OUT]
{
    const int lane = threadIdx.x & 31;
    const int wave = threadIdx.x >> 5;            // 0..7
    const int tile = blockIdx.x * 8 + wave;       // 32768 tiles total
    const int NT   = OUT_ / 16;                   // 16
    const int mt   = tile / NT;
    const int nt   = tile % NT;
    const int col  = lane & 15;
    const int rsel = lane >> 4;

    const __bf16* abase = hsbf + (long)(mt * 16 + col) * H_;
    v8f acc = zero8();
    for (int kc = 0; kc < KCH_O; ++kc) {
        v16bf bfrag = load_b_frag(wop + ((long)(nt * KCH_O + kc) * 32) * 16, lane);
        v16bf afrag = load_a_frag(abase, H_, 0, kc * 32, lane);
        acc = __builtin_amdgcn_wmma_f32_16x16x32_bf16(
            false, afrag, false, bfrag, (short)0, acc, false, false);
    }
    const float bias = bo[nt * 16 + col];
#pragma unroll
    for (int r = 0; r < 8; ++r) {
        const int row = mt * 16 + r + rsel * 8;
        const float v = acc[r] + bias;
        out[(long)row * OUT_ + nt * 16 + col] = v > 0.f ? v : 0.f;
    }
}

// ---------------- host side ----------------
static inline size_t align256(size_t x) { return (x + 255) & ~(size_t)255; }

extern "C" void kernel_launch(void* const* d_in, const int* in_sizes, int n_in,
                              void* d_out, int out_size, void* d_ws, size_t ws_size,
                              hipStream_t stream) {
    (void)in_sizes; (void)n_in; (void)out_size; (void)ws_size;

    const float* x  = (const float*)d_in[0];  // [64][512][256]
    const float* Wf = (const float*)d_in[1];  // [768][512]
    const float* bf = (const float*)d_in[2];  // [512]
    const float* Ws = (const float*)d_in[3];  // [768][512]
    const float* bs = (const float*)d_in[4];  // [512]
    const float* Wo = (const float*)d_in[5];  // [512][256]
    const float* bo = (const float*)d_in[6];  // [256]

    float* out         = (float*)d_out;                       // outputs [64][512][256]
    float* final_state = out + (long)B_ * S_ * OUT_;          // then [64][512]

    char* ws = (char*)d_ws;
    __bf16* xbf  = (__bf16*)ws;  ws += align256((size_t)B_ * S_ * IN_ * 2);
    __bf16* wfp  = (__bf16*)ws;  ws += align256((size_t)KTOT * H_ * 2);
    __bf16* wsp  = (__bf16*)ws;  ws += align256((size_t)KTOT * H_ * 2);
    __bf16* wop  = (__bf16*)ws;  ws += align256((size_t)H_ * OUT_ * 2);
    __bf16* hsbf = (__bf16*)ws;  ws += align256((size_t)B_ * S_ * H_ * 2);

    // 1) convert inputs to bf16
    cvt_bf16_kernel<<<512, 256, 0, stream>>>(x, xbf, (long)B_ * S_ * IN_);
    // 2) pack weights into B-fragment layout (bf16)
    pack_w_kernel<<<256, 256, 0, stream>>>(Wf, wfp, KTOT, H_);
    pack_w_kernel<<<256, 256, 0, stream>>>(Ws, wsp, KTOT, H_);
    pack_w_kernel<<<64,  256, 0, stream>>>(Wo, wop, H_, OUT_);
    // 3) batch-parallel recurrence: 4 independent 16-row chains, 32 KB LDS each
    mgu_rec_kernel<<<4, 1024, 0, stream>>>(xbf, wfp, wsp, bf, bs, hsbf,
                                           final_state);
    // 4) parallel output projection + ReLU
    proj_kernel<<<4096, 256, 0, stream>>>(hsbf, wop, bo, out);
}
